// GATPolicy_multitask_82386062672391
// MI455X (gfx1250) — matrix operand order repaired and checked by reading, not compiled, not run on hardware
//
#include <hip/hip_runtime.h>
#include <hip/hip_bf16.h>
#include <stdint.h>

// ---------------------------------------------------------------------------
// GATPolicy (bipartite GAT, EMB=64, HEADS=8) for MI455X / gfx1250.
//
//  * Dense 64-wide GEMMs: v_wmma_f32_16x16x32_f16. W^T staged once per block
//    into LDS as f16 (K-major) -> B fragments are 16 strided f16 LDS reads per tile.
//    A fragments: float4 (global_load_b128) + v_cvt_pk_f16_f32.
//  * Edge phase (1M edges) is the bottleneck: x_l/x_r/agg tables fit in the
//    192 MB L2, so gathers + global_atomic_add_f32 stay L2-resident. All edge
//    gathers are 32B-aligned float4 loads.
//  * Edge embedding is rank-1 -> recomputed per edge, never materialized.
//
// Input pointer order: setup_inputs() dict insertion order, params flattened
// recursively in insertion order:
//   0 constraint_features[50000,4]   1 edge_indices[2,1000000] (int32)
//   2 edge_features[1000000,1]       3 variable_features[100000,15]
//   4 task_id[1] (int32)
//   5 cons_shift 6 cons_scale 7 cons_W1 8 cons_b1 9 cons_W2 10 cons_b2
//   11 edge_shift 12 edge_scale 13 edge_W 14 edge_b
//   15 var_shift 16 var_scale 17 var_W1 18 var_b1 19 var_W2 20 var_b2
//   21..29  v2c: Wl bl Wr br att Wo1 bo1 Wo2 bo2
//   30..38  c2v: Wl bl Wr br att Wo1 bo1 Wo2 bo2
//   39 out1_W1 40 out1_b1 41 out1_W2 42 out2_W1 43 out2_b1 44 out2_W2
// ---------------------------------------------------------------------------

typedef __attribute__((ext_vector_type(16))) _Float16 v16h;
typedef __attribute__((ext_vector_type(8)))  float    v8f;

#define NEG_SLOPE 0.2f

// ---- ordered-uint encoding of float for atomicMax (memset(0) == -infinity) --
__device__ __forceinline__ unsigned f2ord(float f) {
  unsigned u = __float_as_uint(f);
  return (u & 0x80000000u) ? ~u : (u | 0x80000000u);
}
__device__ __forceinline__ float ord2f(unsigned u) {
  unsigned v = (u & 0x80000000u) ? (u & 0x7FFFFFFFu) : ~u;
  return __uint_as_float(v);
}

// ---------------------------------------------------------------------------
// WMMA GEMM:  Y[M,64] = act( X @ W^T + bias ),  X rows 64-wide, W = [64,K].
// ksteps==2 : K=64, single source X1.
// ksteps==4 : K=128, X row = concat(X1[row], X2[row]).
// Block = 256 threads = 8 waves = 128 rows. W^T staged in LDS as f16 K-major.
// ---------------------------------------------------------------------------
__global__ __launch_bounds__(256)
void gemm64_wmma(const float* __restrict__ X1, const float* __restrict__ X2,
                 const float* __restrict__ W, const float* __restrict__ bias,
                 float* __restrict__ Y, int M, int ksteps, int ldw, int relu) {
  __shared__ _Float16 WhT[128 * 64];          // [k][n], up to 16 KB

  // cooperative transpose W[64][ldw] (f32) -> WhT[ldw][64] (f16)
  for (int i = threadIdx.x; i < 64 * ldw; i += 256) {
    const int n = i / ldw, k = i - n * ldw;   // read coalesced along k
    WhT[k * 64 + n] = (_Float16)W[i];
  }
  __syncthreads();

  const int wave = threadIdx.x >> 5;
  const int lane = threadIdx.x & 31;
  const int row0 = (blockIdx.x * 8 + wave) * 16;
  if (row0 >= M) return;

  v8f c[4];
  #pragma unroll
  for (int t = 0; t < 4; ++t)
    #pragma unroll
    for (int j = 0; j < 8; ++j) c[t][j] = 0.0f;

  const int rloc = lane & 15;
  const int hi   = lane >> 4;
  const long row = row0 + rloc;

  for (int ks = 0; ks < ksteps; ++ks) {
    const float* Xp = (ks < 2) ? X1 : X2;
    const int kl = (ks & 1) * 32;             // K offset within 64-wide source
    const int kw = ks * 32;                   // K offset within W columns
    // A tile (ISA 16-bit A layout): lane holds K[kb..kb+7] and K[kb+16..kb+23]
    const float4* xp = (const float4*)(Xp + row * 64 + kl + hi * 8);
    const float4 x0 = xp[0], x1 = xp[1];      // global_load_b128 x2
    const float4 x2 = xp[4], x3 = xp[5];      // +16 floats
    float xs[16] = {x0.x, x0.y, x0.z, x0.w, x1.x, x1.y, x1.z, x1.w,
                    x2.x, x2.y, x2.z, x2.w, x3.x, x3.y, x3.z, x3.w};
    v16h a;
    #pragma unroll
    for (int j = 0; j < 16; ++j) a[j] = (_Float16)xs[j];

    const int kb = kw + hi * 8;               // B: lane holds K[kb..kb+7], K[kb+16..kb+23] at N = rloc
    #pragma unroll
    for (int nt = 0; nt < 4; ++nt) {
      v16h b;
      #pragma unroll
      for (int j = 0; j < 16; ++j)
        b[j] = WhT[(kb + (j & 7) + (j >> 3) * 16) * 64 + nt * 16 + rloc];
      c[nt] = __builtin_amdgcn_wmma_f32_16x16x32_f16(
          false, a, false, b, (short)0, c[nt], false, false);
    }
  }

  // C/D layout: VGPR r -> M = r + 8*hi, N = (lane&15) + 16*nt
  const int mbase = row0 + hi * 8;
  #pragma unroll
  for (int nt = 0; nt < 4; ++nt) {
    const int n = nt * 16 + rloc;
    const float bv = bias[n];
    #pragma unroll
    for (int r = 0; r < 8; ++r) {
      float v = c[nt][r] + bv;
      if (relu) v = v > 0.0f ? v : 0.0f;
      Y[(long)(mbase + r) * 64 + n] = v;
    }
  }
}

// ---------------------------------------------------------------------------
// First-layer embeddings (tiny K = 4 or 15): y = relu(W @ ((x+shift)*scale)+b)
// ---------------------------------------------------------------------------
__global__ __launch_bounds__(256)
void embed_small(const float* __restrict__ X, int K,
                 const float* __restrict__ shift, const float* __restrict__ scale,
                 const float* __restrict__ W, const float* __restrict__ b,
                 float* __restrict__ Y, int M) {
  const int tid = blockIdx.x * blockDim.x + threadIdx.x;
  if (tid >= M * 64) return;
  const int rowi = tid >> 6, col = tid & 63;
  const float* xr = X + (long)rowi * K;
  const float* wr = W + (long)col * K;
  float acc = b[col];
  for (int k = 0; k < K; ++k) acc += wr[k] * ((xr[k] + shift[k]) * scale[k]);
  Y[tid] = acc > 0.0f ? acc : 0.0f;
}

__device__ __forceinline__ void ld8(const float* p, float* v) {
  const float4* q = (const float4*)p;         // 32B-aligned -> global_load_b128
  const float4 a0 = q[0], a1 = q[1];
  v[0]=a0.x; v[1]=a0.y; v[2]=a0.z; v[3]=a0.w;
  v[4]=a1.x; v[5]=a1.y; v[6]=a1.z; v[7]=a1.w;
}

// ---------------------------------------------------------------------------
// GAT pass 1: per (edge, head) raw attention logit + segment max (ordered u32)
// concat layout [x_i(0..63), x_j(64..127), e(128..191)]; head h owns 8-float
// chunks 3h..3h+2 -> all gathers are float4 pairs.
// ---------------------------------------------------------------------------
__global__ __launch_bounds__(256)
void gat_alpha_max(int NE, const int* __restrict__ src, const int* __restrict__ dst,
                   const float* __restrict__ xl, const float* __restrict__ xr,
                   const float* __restrict__ ef,
                   const float* __restrict__ eshift, const float* __restrict__ escale,
                   const float* __restrict__ eW, const float* __restrict__ eb,
                   const float* __restrict__ att,
                   float* __restrict__ alpha, unsigned* __restrict__ amax) {
  const int tid = blockIdx.x * blockDim.x + threadIdx.x;
  if (tid >= NE * 8) return;
  const int e = tid >> 3, h = tid & 7;
  const int s = src[e], d = dst[e];
  const float t = (ef[e] + eshift[0]) * escale[0];
  const int g0 = h * 24;
  float acc = 0.0f;
  #pragma unroll
  for (int j = 0; j < 3; ++j) {
    const int cch = 3 * h + j;                 // chunk index 0..23
    float v[8];
    if (cch < 8)       ld8(xr + (long)d * 64 + 8 * cch, v);
    else if (cch < 16) ld8(xl + (long)s * 64 + 8 * (cch - 8), v);
    else {
      const int b0 = 8 * (cch - 16);
      #pragma unroll
      for (int k = 0; k < 8; ++k) v[k] = t * eW[b0 + k] + eb[b0 + k];
    }
    #pragma unroll
    for (int k = 0; k < 8; ++k) {
      const float xv = v[k] > 0.0f ? v[k] : NEG_SLOPE * v[k];   // leaky_relu
      acc += xv * att[g0 + 8 * j + k];
    }
  }
  alpha[tid] = acc;
  atomicMax(&amax[(long)d * 8 + h], f2ord(acc));
}

// pass 2: w = exp(a - segmax); denom += w; stash w in alpha buffer
__global__ __launch_bounds__(256)
void gat_expsum(int NE, const int* __restrict__ dst,
                const unsigned* __restrict__ amax, float* __restrict__ denom,
                float* __restrict__ alpha) {
  const int tid = blockIdx.x * blockDim.x + threadIdx.x;
  if (tid >= NE * 8) return;
  const int e = tid >> 3, h = tid & 7;
  const int d = dst[e];
  const float w = __expf(alpha[tid] - ord2f(amax[(long)d * 8 + h]));
  alpha[tid] = w;
  atomicAdd(&denom[(long)d * 8 + h], w);
}

// pass 3: agg[d, 8h..8h+7] += (w / denom) * x_j[s, 8h..8h+7]
__global__ __launch_bounds__(256)
void gat_scatter(int NE, const int* __restrict__ src, const int* __restrict__ dst,
                 const float* __restrict__ xl, const float* __restrict__ denom,
                 const float* __restrict__ alpha, float* __restrict__ agg) {
  const int tid = blockIdx.x * blockDim.x + threadIdx.x;
  if (tid >= NE * 8) return;
  const int e = tid >> 3, h = tid & 7;
  const int s = src[e], d = dst[e];
  float xj[8];
  ld8(xl + (long)s * 64 + h * 8, xj);
  const float w = alpha[tid] / (denom[(long)d * 8 + h] + 1e-16f);
  float* ao = agg + (long)d * 64 + h * 8;
  #pragma unroll
  for (int c = 0; c < 8; ++c) atomicAdd(&ao[c], xj[c] * w);  // global_atomic_add_f32
}

// device-side head-weight selection (task_id lives in device memory)
__global__ __launch_bounds__(256)
void select_head(const int* __restrict__ task,
                 const float* __restrict__ W1a, const float* __restrict__ b1a,
                 const float* __restrict__ w2a,
                 const float* __restrict__ W1b, const float* __restrict__ b1b,
                 const float* __restrict__ w2b, float* __restrict__ dstbuf) {
  const int i = blockIdx.x * blockDim.x + threadIdx.x;
  const bool t1 = (*task == 1);
  if (i < 4096)       dstbuf[i] = t1 ? W1a[i] : W1b[i];
  else if (i < 4160)  dstbuf[i] = t1 ? b1a[i - 4096] : b1b[i - 4096];
  else if (i < 4224)  dstbuf[i] = t1 ? w2a[i - 4160] : w2b[i - 4160];
}

// final head: out[r] = dot(H[r,:64], w2)
__global__ __launch_bounds__(256)
void rowdot64(const float* __restrict__ H, const float* __restrict__ w2,
              float* __restrict__ out, int M) {
  const int r = blockIdx.x * blockDim.x + threadIdx.x;
  if (r >= M) return;
  const float4* hp = (const float4*)(H + (long)r * 64);
  const float4* wp = (const float4*)w2;
  float acc = 0.0f;
  #pragma unroll
  for (int k = 0; k < 16; ++k) {
    const float4 a = hp[k], b = wp[k];
    acc += a.x * b.x + a.y * b.y + a.z * b.z + a.w * b.w;
  }
  out[r] = acc;
}

// ---------------------------------------------------------------------------
extern "C" void kernel_launch(void* const* d_in, const int* in_sizes, int n_in,
                              void* d_out, int out_size, void* d_ws, size_t ws_size,
                              hipStream_t stream) {
  const float* cons_x  = (const float*)d_in[0];
  const int*   eidx    = (const int*)  d_in[1];
  const float* edge_x  = (const float*)d_in[2];
  const float* var_x   = (const float*)d_in[3];
  const int*   task    = (const int*)  d_in[4];
  const float** P = (const float**)(d_in + 5);   // params, insertion order

  const int NC = in_sizes[0] / 4;     // 50000
  const int NE = in_sizes[2];         // 1000000
  const int NV = in_sizes[3] / 15;    // 100000
  const int* cons_idx = eidx;         // edge_indices[0]
  const int* var_idx  = eidx + NE;    // edge_indices[1]

  // workspace layout (floats), total ~41.6M floats (~167 MB)
  float* Wsp = (float*)d_ws;
  float* v0    = Wsp;                       // variable embeddings / v1
  float* c0    = v0   + (long)NV * 64;      // constraint embeddings
  float* xl    = c0   + (long)NC * 64;      // x_l
  float* xr    = xl   + (long)NV * 64;      // x_r
  float* agg   = xr   + (long)NV * 64;      // segment-sum accumulator
  float* c1    = agg  + (long)NV * 64;      // v2c output constraints
  float* amaxb = c1   + (long)NC * 64;      // segment max (ordered u32)
  float* denom = amaxb + (long)NV * 8;      // segment exp-sum
  float* alpha = denom + (long)NV * 8;      // per-(edge,head) buffer
  float* tmp   = alpha;                     // reuse alpha region as MLP temp
  float* headW = alpha + (long)NE * 8;      // selected head weights (4224)

  const int TB = 256;
  const int gC  = (NC * 64 + TB - 1) / TB;
  const int gV  = (NV * 64 + TB - 1) / TB;
  const int gEC = (NC + 127) / 128;
  const int gEV = (NV + 127) / 128;
  const int gE8 = (NE * 8 + TB - 1) / TB;

  // ---- embeddings -----------------------------------------------------------
  embed_small<<<gC, TB, 0, stream>>>(cons_x, 4,  P[0], P[1], P[2], P[3], c1, NC);
  gemm64_wmma<<<gEC, TB, 0, stream>>>(c1, nullptr, P[4], P[5], c0, NC, 2, 64, 1);
  embed_small<<<gV, TB, 0, stream>>>(var_x, 15, P[10], P[11], P[12], P[13], agg, NV);
  gemm64_wmma<<<gEV, TB, 0, stream>>>(agg, nullptr, P[14], P[15], v0, NV, 2, 64, 1);

  // ---- v2c: left = vars (src = var_idx), right = cons (dst = cons_idx) ------
  {
    const float** G = P + 16;  // v2c: Wl bl Wr br att Wo1 bo1 Wo2 bo2
    gemm64_wmma<<<gEV, TB, 0, stream>>>(v0, nullptr, G[0], G[1], xl, NV, 2, 64, 0);
    gemm64_wmma<<<gEC, TB, 0, stream>>>(c0, nullptr, G[2], G[3], xr, NC, 2, 64, 0);
    hipMemsetAsync(amaxb, 0, (long)NC * 8  * 4, stream);  // ordered 0 == -max
    hipMemsetAsync(denom, 0, (long)NC * 8  * 4, stream);
    hipMemsetAsync(agg,   0, (long)NC * 64 * 4, stream);
    gat_alpha_max<<<gE8, TB, 0, stream>>>(NE, var_idx, cons_idx, xl, xr, edge_x,
                                          P[6], P[7], P[8], P[9], G[4],
                                          alpha, (unsigned*)amaxb);
    gat_expsum<<<gE8, TB, 0, stream>>>(NE, cons_idx, (const unsigned*)amaxb, denom, alpha);
    gat_scatter<<<gE8, TB, 0, stream>>>(NE, var_idx, cons_idx, xl, denom, alpha, agg);
    gemm64_wmma<<<gEC, TB, 0, stream>>>(agg, c0, G[5], G[6], tmp, NC, 4, 128, 1);
    gemm64_wmma<<<gEC, TB, 0, stream>>>(tmp, nullptr, G[7], G[8], c1, NC, 2, 64, 0);
  }

  // ---- c2v: left = cons (src = cons_idx), right = vars (dst = var_idx) ------
  {
    const float** G = P + 25;  // c2v: Wl bl Wr br att Wo1 bo1 Wo2 bo2
    gemm64_wmma<<<gEC, TB, 0, stream>>>(c1, nullptr, G[0], G[1], xl, NC, 2, 64, 0);
    gemm64_wmma<<<gEV, TB, 0, stream>>>(v0, nullptr, G[2], G[3], xr, NV, 2, 64, 0);
    hipMemsetAsync(amaxb, 0, (long)NV * 8  * 4, stream);
    hipMemsetAsync(denom, 0, (long)NV * 8  * 4, stream);
    hipMemsetAsync(agg,   0, (long)NV * 64 * 4, stream);
    gat_alpha_max<<<gE8, TB, 0, stream>>>(NE, cons_idx, var_idx, xl, xr, edge_x,
                                          P[6], P[7], P[8], P[9], G[4],
                                          alpha, (unsigned*)amaxb);
    gat_expsum<<<gE8, TB, 0, stream>>>(NE, var_idx, (const unsigned*)amaxb, denom, alpha);
    gat_scatter<<<gE8, TB, 0, stream>>>(NE, cons_idx, var_idx, xl, denom, alpha, agg);
    gemm64_wmma<<<gEV, TB, 0, stream>>>(agg, v0, G[5], G[6], tmp, NV, 4, 128, 1);
    gemm64_wmma<<<gEV, TB, 0, stream>>>(tmp, nullptr, G[7], G[8], v0, NV, 2, 64, 0);
  }

  // ---- output head (task-selected on device) --------------------------------
  select_head<<<(4224 + TB - 1) / TB, TB, 0, stream>>>(
      task, P[34], P[35], P[36], P[37], P[38], P[39], headW);
  gemm64_wmma<<<gEV, TB, 0, stream>>>(v0, nullptr, headW, headW + 4096, tmp, NV, 2, 64, 1);
  rowdot64<<<(NV + TB - 1) / TB, TB, 0, stream>>>(tmp, headW + 4160, (float*)d_out, NV);

  (void)n_in; (void)out_size; (void)ws_size;
}
